// FineGrainedOpGridLSTMNet_47399259079397
// MI455X (gfx1250) — matrix operand order repaired and checked by reading, not compiled, not run on hardware
//
#include <hip/hip_runtime.h>
#include <hip/hip_bf16.h>

typedef __bf16 bf16_t;
typedef bf16_t v16bf __attribute__((ext_vector_type(16)));
typedef float  v8f   __attribute__((ext_vector_type(8)));

#define GS 48      // S
#define GT 48      // T
#define GB 32      // batch (M of cell GEMMs)
#define GH 256     // hidden (N = K = 256)
#define KB 8       // K blocks of 32
#define BH (GB*GH) // 8192 elements per (s,t,c) slab
#define PKE (2*KB*32*16)   // 8192 bf16 elements per packed 32x256 matrix

__device__ __forceinline__ unsigned short f2bf(float x) {
    union { float f; unsigned int u; } v; v.f = x;
    unsigned int u = v.u;
    u += 0x7FFFu + ((u >> 16) & 1u);          // round-to-nearest-even
    return (unsigned short)(u >> 16);
}

// ---- fragment layout helpers ---------------------------------------------
// A (16x32 bf16, MxK): lane m = lane&15; k0 = (lane>=16)?16:0;
//   element e -> k = k0 + e. Packed so each lane's 16
//   elements are contiguous 32B at ((mb*8+kb)*32+lane)*16.
__device__ __forceinline__ v16bf ldA(const unsigned short* base, int mb, int kb, int lane) {
    return *reinterpret_cast<const v16bf*>(base + (((mb * KB) + kb) * 32 + lane) * 16);
}
// B (32x16 bf16, KxN): lane n = lane&15; k = (lane>=16?16:0)+e. Weights are
// stored transposed (N x K, bf16) so each lane reads 16 contiguous bf16 (32B).
__device__ __forceinline__ v16bf ldB(const unsigned short* wt, int nb, int kb, int lane) {
    int n = nb * 16 + (lane & 15);
    int k = kb * 32 + ((lane >> 4) ? 16 : 0);
    return *reinterpret_cast<const v16bf*>(wt + n * GH + k);
}
// inverse of the A-fragment map: flat packed index for element (b_row, h)
__device__ __forceinline__ int packIdx(int b, int h) {
    int mb = b >> 4, lo = b & 15, kb = h >> 5, kk = h & 31;
    int lhi = kk >> 4, e = kk & 15;
    return ((mb * KB + kb) * 32 + lhi * 16 + lo) * 16 + e;
}

__device__ __forceinline__ v8f wmma_bf16(v16bf a, v16bf b, v8f c) {
    return __builtin_amdgcn_wmma_f32_16x16x32_bf16(false, a, false, b, (short)0, c, false, false);
}

// Pack a 32x256 fp32 activation matrix (or zeros if src==nullptr) into LDS
// in the contiguous-k A-fragment layout (k = 16*(lane>>4) + e) (fallback path).
__device__ __forceinline__ void pack_A(const float* src, unsigned short* dst, int tid) {
    for (int f = tid; f < PKE; f += 256) {
        int e  = f & 15;
        int l  = (f >> 4) & 31;
        int kb = (f >> 9) & 7;
        int mb = (f >> 12) & 1;
        int m  = mb * 16 + (l & 15);
        int k  = kb * 32 + ((l >> 4) ? 16 : 0) + e;
        float v = src ? src[m * GH + k] : 0.0f;
        dst[f] = f2bf(v);
    }
}

__device__ __forceinline__ size_t offH(int d, int s, int t, int c) {
    return ((size_t)(((d * GS + s) * GT + t) * 2 + c)) * BH;
}
__device__ __forceinline__ size_t pkOff(int s, int t) {
    return ((size_t)(s * GT + t)) * PKE;
}

// ---- kernel: transpose+cast weights to bf16 (N x K row-major) -------------
// wpack matrix ids: d*3 + {0:W, 1:Ux, 2:Uy}
__global__ __launch_bounds__(256) void pack_weights(const float* __restrict__ W,
                                                    const float* __restrict__ U,
                                                    unsigned short* __restrict__ wpack) {
    int idx = blockIdx.x * 256 + threadIdx.x;
    if (idx >= 6 * GH * GH) return;
    int mat = idx >> 16;
    int rem = idx & 65535;
    int n = rem >> 8;
    int k = rem & 255;
    int d = mat / 3, which = mat % 3;
    float v;
    if (which == 0)      v = W[d * GH * GH + k * GH + n];
    else if (which == 1) v = U[d * 2 * GH * GH + k * GH + n];
    else                 v = U[d * 2 * GH * GH + (k + GH) * GH + n];
    wpack[mat * GH * GH + n * GH + k] = f2bf(v);
}

__global__ __launch_bounds__(256) void zero_u16(unsigned short* __restrict__ p, int n) {
    int i = blockIdx.x * 256 + threadIdx.x;
    if (i < n) p[i] = 0;
}

// ---- kernel: C[R x 256] = A[R x 256] @ W (wt = W^T bf16), 32 rows/block ---
__global__ __launch_bounds__(256) void proj_gemm(const float* __restrict__ A,
                                                 const unsigned short* __restrict__ wt,
                                                 float* __restrict__ C) {
    __shared__ unsigned short sA[PKE];
    int tid = threadIdx.x;
    int rb  = blockIdx.x;
    pack_A(A + (size_t)rb * 32 * GH, sA, tid);
    __syncthreads();

    int wave = tid >> 5, lane = tid & 31;
    int hi = lane >> 4, nl = lane & 15;
    for (int nbi = 0; nbi < 2; ++nbi) {
        int nb = wave * 2 + nbi;
        v8f acc0 = {}, acc1 = {};
        for (int kb = 0; kb < KB; ++kb) {
            v16bf a0 = ldA(sA, 0, kb, lane);
            v16bf a1 = ldA(sA, 1, kb, lane);
            v16bf bw = ldB(wt, nb, kb, lane);
            acc0 = wmma_bf16(a0, bw, acc0);
            acc1 = wmma_bf16(a1, bw, acc1);
        }
        int n = nb * 16 + nl;
        for (int r = 0; r < 8; ++r) {
            int row0 = hi * 8 + r;
            C[((size_t)rb * 32 + row0)      * GH + n] = acc0[r];
            C[((size_t)rb * 32 + 16 + row0) * GH + n] = acc1[r];
        }
    }
}

// ---- fast path: fragments flow cell->cell in packed bf16 form -------------
// No LDS, no barrier: A fragments are per-lane contiguous 32B global loads
// (written by the producing cell's epilogue, L2-resident).
template <bool L1>
__global__ __launch_bounds__(256) void grid_cell_packed(
        float* __restrict__ out, int d, int diag, int smin,
        const float* __restrict__ XW0, const float* __restrict__ YW0,
        const unsigned short* __restrict__ wT,
        const unsigned short* __restrict__ uxT,
        const unsigned short* __restrict__ uyT,
        const float* __restrict__ bias,
        unsigned short* __restrict__ hxpack,        // this layer's packed hx (r/w)
        unsigned short* __restrict__ hypack,        // layer-0 packed hy (write), null for L1
        const unsigned short* __restrict__ xpack,   // L1: layer-0 packed hx
        const unsigned short* __restrict__ ypack,   // L1: layer-0 packed hy
        const unsigned short* __restrict__ zslab) {
    int tid = threadIdx.x;
    int s = smin + blockIdx.x;
    int t = diag - s;

    const unsigned short* upP = (s > 0) ? hxpack + pkOff(s - 1, t) : zslab;
    const unsigned short* lfP = (t > 0) ? hxpack + pkOff(s, t - 1) : zslab;
    const unsigned short* xP  = L1 ? xpack + pkOff(s, t) : nullptr;
    const unsigned short* yP  = L1 ? ypack + pkOff(s, t) : nullptr;

    int wave = tid >> 5, lane = tid & 31;
    int hi = lane >> 4, nl = lane & 15;
    float* hxo = out + offH(d, s, t, 0);
    float* hyo = out + offH(d, s, t, 1);
    unsigned short* hxpo = hxpack + pkOff(s, t);
    unsigned short* hypo = L1 ? nullptr : hypack + pkOff(s, t);

    for (int nbi = 0; nbi < 2; ++nbi) {
        int nb = wave * 2 + nbi;
        v8f t0 = {}, t1 = {};
        v8f x0 = {}, x1 = {}, y0 = {}, y1 = {};
        for (int kb = 0; kb < KB; ++kb) {
            v16bf up0 = ldA(upP, 0, kb, lane);
            v16bf up1 = ldA(upP, 1, kb, lane);
            v16bf lf0 = ldA(lfP, 0, kb, lane);
            v16bf lf1 = ldA(lfP, 1, kb, lane);
            v16bf bux = ldB(uxT, nb, kb, lane);
            v16bf buy = ldB(uyT, nb, kb, lane);
            t0 = wmma_bf16(up0, bux, t0);
            t0 = wmma_bf16(lf0, buy, t0);
            t1 = wmma_bf16(up1, bux, t1);
            t1 = wmma_bf16(lf1, buy, t1);
            if (L1) {
                v16bf xf0 = ldA(xP, 0, kb, lane);
                v16bf xf1 = ldA(xP, 1, kb, lane);
                v16bf yf0 = ldA(yP, 0, kb, lane);
                v16bf yf1 = ldA(yP, 1, kb, lane);
                v16bf bw  = ldB(wT, nb, kb, lane);
                x0 = wmma_bf16(xf0, bw, x0);
                x1 = wmma_bf16(xf1, bw, x1);
                y0 = wmma_bf16(yf0, bw, y0);
                y1 = wmma_bf16(yf1, bw, y1);
            }
        }
        int n = nb * 16 + nl;
        float bv = bias[n];
        for (int r = 0; r < 8; ++r) {
            int row0 = hi * 8 + r;
            int row1 = 16 + row0;
            float tv0 = t0[r] + bv;
            float tv1 = t1[r] + bv;
            float xw0v = L1 ? x0[r] : XW0[((size_t)s * GB + row0) * GH + n];
            float yw0v = L1 ? y0[r] : YW0[((size_t)t * GB + row0) * GH + n];
            float xw1v = L1 ? x1[r] : XW0[((size_t)s * GB + row1) * GH + n];
            float yw1v = L1 ? y1[r] : YW0[((size_t)t * GB + row1) * GH + n];
            float hx0 = tanhf(xw0v + tv0);
            float hy0 = tanhf(yw0v + tv0);
            float hx1 = tanhf(xw1v + tv1);
            float hy1 = tanhf(yw1v + tv1);
            hxo[row0 * GH + n] = hx0;
            hyo[row0 * GH + n] = hy0;
            hxo[row1 * GH + n] = hx1;
            hyo[row1 * GH + n] = hy1;
            hxpo[packIdx(row0, n)] = f2bf(hx0);
            hxpo[packIdx(row1, n)] = f2bf(hx1);
            if (!L1) {
                hypo[packIdx(row0, n)] = f2bf(hy0);
                hypo[packIdx(row1, n)] = f2bf(hy1);
            }
        }
    }
}

// ---- fallback path (small workspace): LDS staging as in round 1 -----------
template <bool L1>
__global__ __launch_bounds__(256) void grid_cell_lds(
        float* __restrict__ out, int d, int diag, int smin,
        const float* __restrict__ XW0, const float* __restrict__ YW0,
        const unsigned short* __restrict__ wT,
        const unsigned short* __restrict__ uxT,
        const unsigned short* __restrict__ uyT,
        const float* __restrict__ bias) {
    __shared__ unsigned short sA[(L1 ? 4 : 2) * PKE];
    int tid = threadIdx.x;
    int s = smin + blockIdx.x;
    int t = diag - s;

    const float* upsrc   = (s > 0) ? out + offH(d, s - 1, t, 0) : nullptr;
    const float* leftsrc = (t > 0) ? out + offH(d, s, t - 1, 0) : nullptr;
    pack_A(upsrc,   sA,       tid);
    pack_A(leftsrc, sA + PKE, tid);
    if (L1) {
        pack_A(out + offH(0, s, t, 0), sA + 2 * PKE, tid);
        pack_A(out + offH(0, s, t, 1), sA + 3 * PKE, tid);
    }
    __syncthreads();

    int wave = tid >> 5, lane = tid & 31;
    int hi = lane >> 4, nl = lane & 15;
    float* hxo = out + offH(d, s, t, 0);
    float* hyo = out + offH(d, s, t, 1);

    for (int nbi = 0; nbi < 2; ++nbi) {
        int nb = wave * 2 + nbi;
        v8f t0 = {}, t1 = {};
        v8f x0 = {}, x1 = {}, y0 = {}, y1 = {};
        for (int kb = 0; kb < KB; ++kb) {
            v16bf up0 = ldA(sA, 0, kb, lane);
            v16bf up1 = ldA(sA, 1, kb, lane);
            v16bf lf0 = ldA(sA + PKE, 0, kb, lane);
            v16bf lf1 = ldA(sA + PKE, 1, kb, lane);
            v16bf bux = ldB(uxT, nb, kb, lane);
            v16bf buy = ldB(uyT, nb, kb, lane);
            t0 = wmma_bf16(up0, bux, t0);
            t0 = wmma_bf16(lf0, buy, t0);
            t1 = wmma_bf16(up1, bux, t1);
            t1 = wmma_bf16(lf1, buy, t1);
            if (L1) {
                v16bf xf0 = ldA(sA + 2 * PKE, 0, kb, lane);
                v16bf xf1 = ldA(sA + 2 * PKE, 1, kb, lane);
                v16bf yf0 = ldA(sA + 3 * PKE, 0, kb, lane);
                v16bf yf1 = ldA(sA + 3 * PKE, 1, kb, lane);
                v16bf bw  = ldB(wT, nb, kb, lane);
                x0 = wmma_bf16(xf0, bw, x0);
                x1 = wmma_bf16(xf1, bw, x1);
                y0 = wmma_bf16(yf0, bw, y0);
                y1 = wmma_bf16(yf1, bw, y1);
            }
        }
        int n = nb * 16 + nl;
        float bv = bias[n];
        for (int r = 0; r < 8; ++r) {
            int row0 = hi * 8 + r;
            int row1 = 16 + row0;
            float tv0 = t0[r] + bv;
            float tv1 = t1[r] + bv;
            float xw0v = L1 ? x0[r] : XW0[((size_t)s * GB + row0) * GH + n];
            float yw0v = L1 ? y0[r] : YW0[((size_t)t * GB + row0) * GH + n];
            float xw1v = L1 ? x1[r] : XW0[((size_t)s * GB + row1) * GH + n];
            float yw1v = L1 ? y1[r] : YW0[((size_t)t * GB + row1) * GH + n];
            hxo[row0 * GH + n] = tanhf(xw0v + tv0);
            hyo[row0 * GH + n] = tanhf(yw0v + tv0);
            hxo[row1 * GH + n] = tanhf(xw1v + tv1);
            hyo[row1 * GH + n] = tanhf(yw1v + tv1);
        }
    }
}

extern "C" void kernel_launch(void* const* d_in, const int* in_sizes, int n_in,
                              void* d_out, int out_size, void* d_ws, size_t ws_size,
                              hipStream_t stream) {
    const float* src = (const float*)d_in[0];   // (48,32,256)
    const float* trg = (const float*)d_in[1];   // (48,32,256)
    const float* W   = (const float*)d_in[2];   // (2,256,256)
    const float* U   = (const float*)d_in[3];   // (2,512,256)
    const float* b   = (const float*)d_in[4];   // (2,1,256)
    float* out = (float*)d_out;

    // workspace layout
    const size_t wpackB = (size_t)6 * GH * GH * 2;            // 768 KB
    const size_t projB  = (size_t)GS * GB * GH * 4;           // 1.5 MB each
    const size_t cellPB = (size_t)GS * GT * PKE * 2;          // 36 MB each
    unsigned short* wpack = (unsigned short*)d_ws;
    float* XW0 = (float*)((char*)d_ws + wpackB);
    float* YW0 = (float*)((char*)XW0 + projB);
    char* extra = (char*)YW0 + projB;
    unsigned short* zslab = (unsigned short*)extra;
    unsigned short* hx0p  = (unsigned short*)(extra + PKE * 2);
    unsigned short* hy0p  = (unsigned short*)((char*)hx0p + cellPB);
    unsigned short* hx1p  = (unsigned short*)((char*)hy0p + cellPB);
    const size_t needPacked = wpackB + 2 * projB + PKE * 2 + 3 * cellPB;
    const bool usePacked = ws_size >= needPacked;

    pack_weights<<<(6 * GH * GH + 255) / 256, 256, 0, stream>>>(W, U, wpack);
    // layer-0 input projections: XW0[s] = src[s]@W0, YW0[t] = trg[t]@W0
    proj_gemm<<<GS, 256, 0, stream>>>(src, wpack + 0 * GH * GH, XW0);
    proj_gemm<<<GT, 256, 0, stream>>>(trg, wpack + 0 * GH * GH, YW0);

    if (usePacked) {
        zero_u16<<<(PKE + 255) / 256, 256, 0, stream>>>(zslab, PKE);
        for (int k = 0; k < GS + GT - 1; ++k) {
            int smin = (k > GT - 1) ? k - (GT - 1) : 0;
            int smax = (k < GS - 1) ? k : GS - 1;
            int cnt = smax - smin + 1;
            grid_cell_packed<false><<<cnt, 256, 0, stream>>>(out, 0, k, smin, XW0, YW0,
                    nullptr, wpack + 1 * GH * GH, wpack + 2 * GH * GH, b + 0 * GH,
                    hx0p, hy0p, nullptr, nullptr, zslab);
        }
        for (int k = 0; k < GS + GT - 1; ++k) {
            int smin = (k > GT - 1) ? k - (GT - 1) : 0;
            int smax = (k < GS - 1) ? k : GS - 1;
            int cnt = smax - smin + 1;
            grid_cell_packed<true><<<cnt, 256, 0, stream>>>(out, 1, k, smin, nullptr, nullptr,
                    wpack + 3 * GH * GH, wpack + 4 * GH * GH, wpack + 5 * GH * GH, b + 1 * GH,
                    hx1p, nullptr, hx0p, hy0p, zslab);
        }
    } else {
        for (int k = 0; k < GS + GT - 1; ++k) {
            int smin = (k > GT - 1) ? k - (GT - 1) : 0;
            int smax = (k < GS - 1) ? k : GS - 1;
            int cnt = smax - smin + 1;
            grid_cell_lds<false><<<cnt, 256, 0, stream>>>(out, 0, k, smin, XW0, YW0,
                    nullptr, wpack + 1 * GH * GH, wpack + 2 * GH * GH, b + 0 * GH);
        }
        for (int k = 0; k < GS + GT - 1; ++k) {
            int smin = (k > GT - 1) ? k - (GT - 1) : 0;
            int smax = (k < GS - 1) ? k : GS - 1;
            int cnt = smax - smin + 1;
            grid_cell_lds<true><<<cnt, 256, 0, stream>>>(out, 1, k, smin, nullptr, nullptr,
                    wpack + 3 * GH * GH, wpack + 4 * GH * GH, wpack + 5 * GH * GH, b + 1 * GH);
        }
    }
}
